// DiffusionWaveGAN_9474697855453
// MI455X (gfx1250) — matrix operand order repaired and checked by reading, not compiled, not run on hardware
//
#include <hip/hip_runtime.h>
#include <math.h>

// ---------------- model constants ----------------
#define T_LEN  16384
#define NBLK   30
#define NPACK  45056   // 88 fragments * 512 halfs per wavenet block

typedef __attribute__((ext_vector_type(16))) _Float16 v16h;
typedef __attribute__((ext_vector_type(8)))  float    v8f;

// ---------------------------------------------------------------------------
// WMMA fragment layouts (CDNA5 ISA 7.12.2, wave32):
//  A (16x32 f16):  lane L: m=L%16, h=L/16; elem e=2i+j:
//      k = 16h+2i+j (contiguous, as packed by k_pack)
//  B (32x16 f16):  lane L: n=L%16, h=L/16; elem e: k = 16h + e (this file's order, not the ISA table order)
//  C (16x16 f32):  vgpr r, lane L: n=L%16, m = r + 8*(L/16)
// ---------------------------------------------------------------------------

// ---------------- step embedding -> e2[step][512] ----------------
__global__ void k_embed(const float* __restrict__ ew0, const float* __restrict__ eb0,
                        const float* __restrict__ ew1, const float* __restrict__ eb1,
                        float* __restrict__ embed_out)
{
    __shared__ float e0[128];
    __shared__ float e1[512];
    const int step = blockIdx.x;          // 0 or 1
    const int t = threadIdx.x;            // 512 threads
    if (t < 128) {
        int i = t & 63;
        float f = expf(-logf(10000.f) * (float)i / 64.f);
        float ang = (float)step * f;
        e0[t] = (t < 64) ? sinf(ang) : cosf(ang);
    }
    __syncthreads();
    float s = eb0[t];
    for (int i = 0; i < 128; i++) s += e0[i] * ew0[i * 512 + t];
    e1[t] = s / (1.f + expf(-s));         // SiLU
    __syncthreads();
    float s2 = eb1[t];
    for (int i = 0; i < 512; i++) s2 += e1[i] * ew1[i * 512 + t];
    embed_out[step * 512 + t] = s2 / (1.f + expf(-s2));
}

// ---------------- per-block weight pack (f32 -> f16 A-fragments) + PE bias ----
__global__ void k_pack(const float* __restrict__ conv_w, const float* __restrict__ mel_w,
                       const float* __restrict__ rs_w,
                       const float* __restrict__ pe_w, const float* __restrict__ pe_b,
                       const float* __restrict__ embed,
                       _Float16* __restrict__ wpack, float* __restrict__ pe_out)
{
    int gid = blockIdx.x * blockDim.x + threadIdx.x;
    if (gid < NPACK) {
        int f = gid >> 9;
        int r = gid & 511;
        int lane = r >> 4, elem = r & 15;
        int h = lane >> 4, ml = lane & 15;
        int i = elem >> 1, jj = elem & 1;
        int klocal = 16 * h + 2 * i + jj;
        float val;
        if (f < 48) {                       // dilated conv taps: [tap][mtile][kc]
            int tap = f / 16, rem = f % 16;
            int mtile = rem >> 1, kc = rem & 1;
            int m = mtile * 16 + ml, k = kc * 32 + klocal;
            val = conv_w[(m * 64 + k) * 3 + tap];          // [128][64][3]
        } else if (f < 72) {                // mel conv, K padded 80 -> 96
            int f2 = f - 48;
            int mtile = f2 / 3, kc = f2 % 3;
            int m = mtile * 16 + ml, k = kc * 32 + klocal;
            val = (k < 80) ? mel_w[m * 80 + k] : 0.f;       // [128][80][1]
        } else {                            // res/skip 1x1
            int f3 = f - 72;
            int mtile = f3 >> 1, kc = f3 & 1;
            int m = mtile * 16 + ml, k = kc * 32 + klocal;
            val = rs_w[m * 64 + k];                         // [128][64][1]
        }
        wpack[gid] = (_Float16)val;
    } else if (gid < NPACK + 128) {         // pe_bias for both steps
        int g2 = gid - NPACK;
        int step = g2 >> 6, c = g2 & 63;
        float s = pe_b[c];
        const float* e = embed + step * 512;
        for (int i = 0; i < 512; i++) s += e[i] * pe_w[i * 64 + c];
        pe_out[step * (NBLK * 64) + c] = s; // pe_out already offset by j*64
    }
}

// ---------------- latent projection chain (1->16, 16->16 k5 x2, 16->64) -----
__global__ __launch_bounds__(256)
void k_latent(const float* __restrict__ latent,
              const float* __restrict__ w0, const float* __restrict__ b0,
              const float* __restrict__ w1, const float* __restrict__ b1,
              const float* __restrict__ w2, const float* __restrict__ b2,
              const float* __restrict__ wo, const float* __restrict__ bo,
              float* __restrict__ latproj)
{
    __shared__ float h0[16][264];
    __shared__ float h1[16][260];
    __shared__ float h2[16][256];
    const int b = blockIdx.y;
    const int t0 = blockIdx.x * 256;
    const int tid = threadIdx.x;

    for (int idx = tid; idx < 16 * 264; idx += 256) {
        int c = idx / 264, p = idx % 264;
        int g = t0 - 4 + p;
        float v = 0.f;
        if (g >= 0 && g < T_LEN) {
            float s = w0[c] * latent[b * T_LEN + g] + b0[c];
            v = s / (1.f + expf(-s));
        }
        h0[c][p] = v;
    }
    __syncthreads();
    for (int idx = tid; idx < 16 * 260; idx += 256) {
        int c = idx / 260, q = idx % 260;
        int g = t0 - 2 + q;
        float v = 0.f;
        if (g >= 0 && g < T_LEN) {
            float s = b1[c];
            for (int ci = 0; ci < 16; ci++)
                for (int k = 0; k < 5; k++) s += w1[(c * 16 + ci) * 5 + k] * h0[ci][q + k];
            v = s / (1.f + expf(-s));
        }
        h1[c][q] = v;
    }
    __syncthreads();
    for (int idx = tid; idx < 16 * 256; idx += 256) {
        int c = idx / 256, q = idx % 256;
        float s = b2[c];
        for (int ci = 0; ci < 16; ci++)
            for (int k = 0; k < 5; k++) s += w2[(c * 16 + ci) * 5 + k] * h1[ci][q + k];
        h2[c][q] = s / (1.f + expf(-s));
    }
    __syncthreads();
    for (int idx = tid; idx < 64 * 256; idx += 256) {
        int co = idx / 256, q = idx % 256;
        float s = bo[co];
        for (int ci = 0; ci < 16; ci++) s += wo[co * 16 + ci] * h2[ci][q];
        latproj[(b * 64 + co) * T_LEN + t0 + q] = s;
    }
}

// ---------------- transposed-conv upsampler stage (stride 4, K=8, pad=5) -----
__global__ void k_up(const float* __restrict__ in, const float* __restrict__ w,
                     const float* __restrict__ bias, float* __restrict__ out,
                     int Tin, int Tout, int src_mel)
{
    int gid = blockIdx.x * blockDim.x + threadIdx.x;
    if (gid >= 2 * 80 * Tout) return;
    int t = gid % Tout;
    int o = (gid / Tout) % 80;
    int b = gid / (Tout * 80);
    float acc = bias[o];
    for (int k = 0; k < 8; k++) {
        int j = t + k - 5;
        if (j < 0 || (j & 3) != 0) continue;
        int j4 = j >> 2;
        if (j4 >= Tin) continue;
        const float* wr = w + (o * 80) * 8 + k;     // w[o][i][k]
        if (src_mel) {                               // mel input: [B][T_MEL][80]
            const float* xin = in + (size_t)(b * Tin) * 80;
            for (int i = 0; i < 80; i++) acc += xin[j4 * 80 + i] * wr[i * 8];
        } else {                                     // [B][80][Tin]
            const float* xin = in + (size_t)(b * 80) * Tin;
            for (int i = 0; i < 80; i++) acc += xin[i * Tin + j4] * wr[i * 8];
        }
    }
    out[gid] = (acc >= 0.f) ? acc : 0.2f * acc;      // LeakyReLU
}

// ---------------- per-step init: x = sig_proj(signal) + latproj; skip = 0 ----
__global__ void k_init(const float* __restrict__ sig, const float* __restrict__ sw,
                       const float* __restrict__ sb, const float* __restrict__ latproj,
                       float* __restrict__ x, float* __restrict__ skipb)
{
    int gid = blockIdx.x * blockDim.x + threadIdx.x;
    if (gid >= 2 * 64 * T_LEN) return;
    int t = gid % T_LEN;
    int c = (gid / T_LEN) % 64;
    int b = gid / (T_LEN * 64);
    x[gid] = sw[c] * sig[b * T_LEN + t] + sb[c] + latproj[gid];
    skipb[gid] = 0.f;
}

// ---------------- WaveNet block: WMMA GEMMs ---------------------------------
struct PanelA { _Float16 bx[3][2][4][512]; _Float16 bmel[3][4][512]; };
union Smem0 { PanelA a; float y[128][64]; };

__global__ __launch_bounds__(256)
void k_block(const float* __restrict__ x_in, float* __restrict__ x_out,
             float* __restrict__ skipb, const float* __restrict__ mel_up,
             const _Float16* __restrict__ wp,
             const float* __restrict__ conv_b, const float* __restrict__ mel_b,
             const float* __restrict__ rs_b, const float* __restrict__ pe_bias,
             int dil)
{
    __shared__ __align__(32) Smem0 u;                 // 36 KB (panels) / 32 KB (y)
    __shared__ __align__(32) float    sXC[64][64];    // 16 KB, x+pe center (f32)
    __shared__ __align__(32) _Float16 sBh[2][4][512]; // 8 KB, gated h fragments

    const int b  = blockIdx.y;
    const int t0 = blockIdx.x * 64;
    const int tid = threadIdx.x;

    // ---- stage shifted x panels (B-fragment f16 layout) + mel panel ----
    for (int idx = tid; idx < 3 * 64 * 64; idx += 256) {
        int tap = idx >> 12;
        int rem = idx & 4095;
        int c = rem >> 6, tl = rem & 63;
        int gt = t0 + tl + (tap - 1) * dil;
        float v = 0.f;
        if (gt >= 0 && gt < T_LEN) v = x_in[(b * 64 + c) * T_LEN + gt] + pe_bias[c];
        int kc = c >> 5, kl = c & 31;
        int nt = tl >> 4, n = tl & 15;
        u.a.bx[tap][kc][nt][(n + ((kl >> 4) << 4)) * 16 + (kl & 15)] = (_Float16)v;
        if (tap == 1) sXC[c][tl] = v;
    }
    for (int idx = tid; idx < 96 * 64; idx += 256) {
        int c = idx >> 6, tl = idx & 63;
        float v = (c < 80) ? mel_up[(b * 80 + c) * T_LEN + t0 + tl] : 0.f;
        int kc = c >> 5, kl = c & 31;
        int nt = tl >> 4, n = tl & 15;
        u.a.bmel[kc][nt][(n + ((kl >> 4) << 4)) * 16 + (kl & 15)] = (_Float16)v;
    }
    __syncthreads();

    const int w    = tid >> 5;    // wave id: M-tile
    const int lane = tid & 31;
    const int hh   = lane >> 4;
    const int nl   = lane & 15;

    v8f acc[4];
    #pragma unroll
    for (int nt = 0; nt < 4; nt++)
        #pragma unroll
        for (int r = 0; r < 8; r++) {
            int m = 16 * w + r + 8 * hh;
            acc[nt][r] = conv_b[m] + mel_b[m];
        }

    // ---- y = dilated conv (3 taps, K=64) + mel conv (K=96) ----
    #pragma unroll
    for (int tap = 0; tap < 3; tap++)
        #pragma unroll
        for (int kc = 0; kc < 2; kc++) {
            v16h afrag = *(const v16h*)(wp + ((((tap * 8 + w) * 2) + kc) << 9) + lane * 16);
            #pragma unroll
            for (int nt = 0; nt < 4; nt++) {
                v16h bfrag = *(const v16h*)(&u.a.bx[tap][kc][nt][0] + lane * 16);
                acc[nt] = __builtin_amdgcn_wmma_f32_16x16x32_f16(
                    false, afrag, false, bfrag, (short)0, acc[nt], false, false);
            }
        }
    #pragma unroll
    for (int kc = 0; kc < 3; kc++) {
        v16h afrag = *(const v16h*)(wp + ((48 + w * 3 + kc) << 9) + lane * 16);
        #pragma unroll
        for (int nt = 0; nt < 4; nt++) {
            v16h bfrag = *(const v16h*)(&u.a.bmel[kc][nt][0] + lane * 16);
            acc[nt] = __builtin_amdgcn_wmma_f32_16x16x32_f16(
                false, afrag, false, bfrag, (short)0, acc[nt], false, false);
        }
    }

    __syncthreads();   // all waves done reading u.a before overwriting with y
    #pragma unroll
    for (int nt = 0; nt < 4; nt++)
        #pragma unroll
        for (int r = 0; r < 8; r++)
            u.y[16 * w + r + 8 * hh][nt * 16 + nl] = acc[nt][r];
    __syncthreads();

    // ---- gating: h = tanh(a) * sigmoid(g), stored as B-fragments ----
    for (int idx = tid; idx < 64 * 64; idx += 256) {
        int c = idx >> 6, tl = idx & 63;
        float aa = u.y[c][tl];
        float gg = u.y[c + 64][tl];
        float hv = tanhf(aa) * (1.f / (1.f + expf(-gg)));
        int kc = c >> 5, kl = c & 31;
        int nt = tl >> 4, n = tl & 15;
        sBh[kc][nt][(n + ((kl >> 4) << 4)) * 16 + (kl & 15)] = (_Float16)hv;
    }
    __syncthreads();

    // ---- res/skip 1x1 conv (K=64) ----
    v8f acc2[4];
    #pragma unroll
    for (int nt = 0; nt < 4; nt++)
        #pragma unroll
        for (int r = 0; r < 8; r++) acc2[nt][r] = rs_b[16 * w + r + 8 * hh];
    #pragma unroll
    for (int kc = 0; kc < 2; kc++) {
        v16h afrag = *(const v16h*)(wp + ((72 + w * 2 + kc) << 9) + lane * 16);
        #pragma unroll
        for (int nt = 0; nt < 4; nt++) {
            v16h bfrag = *(const v16h*)(&sBh[kc][nt][0] + lane * 16);
            acc2[nt] = __builtin_amdgcn_wmma_f32_16x16x32_f16(
                false, afrag, false, bfrag, (short)0, acc2[nt], false, false);
        }
    }

    #pragma unroll
    for (int nt = 0; nt < 4; nt++)
        #pragma unroll
        for (int r = 0; r < 8; r++) {
            int m = 16 * w + r + 8 * hh;
            int gt = t0 + nt * 16 + nl;
            float v = acc2[nt][r];
            if (m < 64) {
                x_out[(b * 64 + m) * T_LEN + gt] = (sXC[m][nt * 16 + nl] + v) * 0.70710678118654752f;
            } else {
                float* sp = &skipb[(b * 64 + (m - 64)) * T_LEN + gt];
                *sp = *sp + v;
            }
        }
}

// ---------------- output head + diffusion update -----------------------------
__global__ __launch_bounds__(256)
void k_out(const float* __restrict__ skipb, const float* __restrict__ w0,
           const float* __restrict__ b0, const float* __restrict__ w1,
           const float* __restrict__ b1, const float* __restrict__ sig_in,
           float* __restrict__ out, float c1, float c2)
{
    __shared__ float sw0[4096];
    __shared__ float sw1[64];
    __shared__ float sb0[64];
    const int tid = threadIdx.x;
    for (int i = tid; i < 4096; i += 256) sw0[i] = w0[i];
    if (tid < 64) { sw1[tid] = w1[tid]; sb0[tid] = b0[tid]; }
    __syncthreads();
    int gid = blockIdx.x * 256 + tid;
    if (gid >= 2 * T_LEN) return;
    int b = gid / T_LEN, t = gid % T_LEN;
    const float scale = 0.18257418583505537f;   // 30^-0.5
    float y[64];
    #pragma unroll
    for (int c = 0; c < 64; c++) {
        float v = skipb[(b * 64 + c) * T_LEN + t] * scale;
        y[c] = (v > 0.f) ? v : 0.f;
    }
    float d = b1[0];
    for (int co = 0; co < 64; co++) {
        float s = sb0[co];
        #pragma unroll
        for (int c = 0; c < 64; c++) s += sw0[co * 64 + c] * y[c];
        s = (s > 0.f) ? s : 0.f;
        d += sw1[co] * s;
    }
    out[gid] = c1 * d + c2 * sig_in[gid];
}

// ---------------------------------------------------------------------------
extern "C" void kernel_launch(void* const* d_in, const int* in_sizes, int n_in,
                              void* d_out, int out_size, void* d_ws, size_t ws_size,
                              hipStream_t stream)
{
    (void)in_sizes; (void)n_in; (void)out_size; (void)ws_size;
    const float* mel    = (const float*)d_in[0];
    const float* signal = (const float*)d_in[1];
    const float* latent = (const float*)d_in[2];
    // params flattened (JAX pytree, sorted keys): blocks[30]{conv_b,conv_w,mel_b,
    // mel_w,pe_b,pe_w,rs_b,rs_w} then emb_b0,emb_b1,emb_w0,emb_w1, lat_b0..2,
    // lat_bo, lat_w0..2, lat_wo, out_b0,out_b1,out_w0,out_w1, sig_b,sig_w,
    // up_b0..3, up_w0..3
    auto P = [&](int i) { return (const float*)d_in[i]; };
    #define BLKP(j,k) ((const float*)d_in[3 + (j) * 8 + (k)])
    const int I_EMB_B0 = 243, I_EMB_B1 = 244, I_EMB_W0 = 245, I_EMB_W1 = 246;
    const int I_LAT_B0 = 247, I_LAT_B1 = 248, I_LAT_B2 = 249, I_LAT_BO = 250;
    const int I_LAT_W0 = 251, I_LAT_W1 = 252, I_LAT_W2 = 253, I_LAT_WO = 254;
    const int I_OUT_B0 = 255, I_OUT_B1 = 256, I_OUT_W0 = 257, I_OUT_W1 = 258;
    const int I_SIG_B  = 259, I_SIG_W  = 260;
    const int I_UP_B   = 261, I_UP_W   = 265;

    // ---- workspace carve ----
    char* ws = (char*)d_ws;
    size_t off = 0;
    auto carve = [&](size_t bytes) {
        void* p = ws + off;
        off = (off + bytes + 255) & ~(size_t)255;
        return p;
    };
    const size_t XSZ = (size_t)2 * 64 * T_LEN;
    float* x_a     = (float*)carve(XSZ * 4);
    float* x_b     = (float*)carve(XSZ * 4);
    float* skipb   = (float*)carve(XSZ * 4);
    float* latproj = (float*)carve(XSZ * 4);
    float* melup   = (float*)carve((size_t)2 * 80 * T_LEN * 4);
    float* u1      = (float*)carve((size_t)2 * 80 * 256 * 4);
    float* u2      = (float*)carve((size_t)2 * 80 * 1024 * 4);
    float* u3      = (float*)carve((size_t)2 * 80 * 4096 * 4);
    float* sigbuf  = (float*)carve((size_t)2 * T_LEN * 4);
    float* embed   = (float*)carve(2 * 512 * 4);
    float* peb     = (float*)carve(2 * NBLK * 64 * 4);
    _Float16* wpack = (_Float16*)carve((size_t)NBLK * NPACK * 2);

    // ---- step-invariant precompute ----
    k_embed<<<dim3(2), dim3(512), 0, stream>>>(P(I_EMB_W0), P(I_EMB_B0),
                                               P(I_EMB_W1), P(I_EMB_B1), embed);
    for (int j = 0; j < NBLK; j++) {
        k_pack<<<dim3((NPACK + 128 + 255) / 256), dim3(256), 0, stream>>>(
            BLKP(j, 1), BLKP(j, 3), BLKP(j, 7), BLKP(j, 5), BLKP(j, 4),
            embed, wpack + (size_t)j * NPACK, peb + j * 64);
    }
    k_latent<<<dim3(T_LEN / 256, 2), dim3(256), 0, stream>>>(
        latent, P(I_LAT_W0), P(I_LAT_B0), P(I_LAT_W1), P(I_LAT_B1),
        P(I_LAT_W2), P(I_LAT_B2), P(I_LAT_WO), P(I_LAT_BO), latproj);
    {   // upsampler: 64 -> 256 -> 1024 -> 4096 -> 16384
        int tin = 64;
        float* bufs[5] = {nullptr, u1, u2, u3, melup};
        const float* src = mel;
        for (int s = 0; s < 4; s++) {
            int tout = tin * 4;
            int total = 2 * 80 * tout;
            k_up<<<dim3((total + 255) / 256), dim3(256), 0, stream>>>(
                src, P(I_UP_W + s), P(I_UP_B + s), bufs[s + 1], tin, tout, s == 0);
            src = bufs[s + 1];
            tin = tout;
        }
    }

    // ---- diffusion coefficients: betas = [0, 0.25, 0.5] ----
    float betas[3] = {0.f, 0.25f, 0.5f};
    float alphas[3], abar[3];
    float cp = 1.f;
    for (int i = 0; i < 3; i++) { alphas[i] = 1.f - betas[i]; cp *= alphas[i]; abar[i] = cp; }
    float c1s[2], c2s[2];
    for (int s = 0; s < 2; s++) {
        int nxt = s + 1;
        c1s[s] = sqrtf(abar[s]) * betas[nxt] / (1.f - abar[nxt]);
        c2s[s] = sqrtf(alphas[nxt]) * (1.f - abar[s]) / (1.f - abar[nxt]);
    }

    // ---- sampling loop: step = 1, then step = 0 ----
    const float* sig_src = signal;
    for (int si = 0; si < 2; si++) {
        int step = 1 - si;
        k_init<<<dim3((int)(XSZ / 256)), dim3(256), 0, stream>>>(
            sig_src, P(I_SIG_W), P(I_SIG_B), latproj, x_a, skipb);
        float* xin = x_a;
        float* xout = x_b;
        for (int j = 0; j < NBLK; j++) {
            int dil = 1 << (j % 10);
            k_block<<<dim3(T_LEN / 64, 2), dim3(256), 0, stream>>>(
                xin, xout, skipb, melup, wpack + (size_t)j * NPACK,
                BLKP(j, 0), BLKP(j, 2), BLKP(j, 6),
                peb + step * (NBLK * 64) + j * 64, dil);
            float* tmp = xin; xin = xout; xout = tmp;
        }
        float* out_sig = (step == 0) ? (float*)d_out : sigbuf;
        k_out<<<dim3((2 * T_LEN + 255) / 256), dim3(256), 0, stream>>>(
            skipb, P(I_OUT_W0), P(I_OUT_B0), P(I_OUT_W1), P(I_OUT_B1),
            sig_src, out_sig, c1s[step], c2s[step]);
        sig_src = sigbuf;
    }
    #undef BLKP
}
